// OuterInterpolatedIPAttnProcessor_20005957665461
// MI455X (gfx1250) — compile-verified, matrix operand order checked by reading, not run on hardware
//
#include <hip/hip_runtime.h>
#include <hip/hip_bf16.h>

// ---------------------------------------------------------------------------
// Types / helpers for CDNA5 WMMA (wave32, v_wmma_f32_16x16x32_bf16)
// ---------------------------------------------------------------------------
typedef __bf16 v16bf __attribute__((ext_vector_type(16)));
typedef float  v8f   __attribute__((ext_vector_type(8)));
typedef int    v4i   __attribute__((ext_vector_type(4)));
typedef __attribute__((address_space(1))) v4i g_v4i;   // global (prints as __device__)
typedef __attribute__((address_space(3))) v4i l_v4i;   // LDS    (prints as __shared__)

#if __has_builtin(__builtin_amdgcn_global_load_async_to_lds_b128) && \
    __has_builtin(__builtin_amdgcn_s_wait_asynccnt)
#define HAVE_ASYNC_LDS 1
#else
#define HAVE_ASYNC_LDS 0
#endif

__device__ __forceinline__ unsigned short f2bf(float f) {
    unsigned int u = __float_as_uint(f);
    u += 0x7FFFu + ((u >> 16) & 1u);          // round-to-nearest-even
    return (unsigned short)(u >> 16);
}

__device__ __forceinline__ v8f v8zero() {
    v8f r;
#pragma unroll
    for (int i = 0; i < 8; ++i) r[i] = 0.0f;
    return r;
}

// 16-byte global -> LDS copy. Async (ASYNCcnt path) when available.
__device__ __forceinline__ void cp16(unsigned short* lds, const unsigned short* g) {
#if HAVE_ASYNC_LDS
    __builtin_amdgcn_global_load_async_to_lds_b128((g_v4i*)g, (l_v4i*)lds, 0, 0);
#else
    *(uint4*)lds = *(const uint4*)g;
#endif
}

// Wait for this wave's async LDS writes, then workgroup barrier.
__device__ __forceinline__ void lds_sync() {
#if HAVE_ASYNC_LDS
    __builtin_amdgcn_s_wait_asynccnt(0);
#endif
    __syncthreads();
}

// Load one 16x32 bf16 WMMA fragment (A or Bt layout) from an LDS row.
__device__ __forceinline__ v16bf ld_frag(const unsigned short* p, int khalf) {
    union { uint4 u[2]; v16bf v; } x;
    x.u[0] = *(const uint4*)(p + khalf * 8);
    x.u[1] = *(const uint4*)(p + 16 + khalf * 8);
    return x.v;
}

__device__ __forceinline__ v8f wmma_bf16(v16bf a, v16bf b, v8f c) {
    return __builtin_amdgcn_wmma_f32_16x16x32_bf16(false, a, false, b, (short)0, c,
                                                   false, false);
}

// ---------------------------------------------------------------------------
// Problem constants
// ---------------------------------------------------------------------------
#define BATCH   7
#define SQ      4096
#define CDIM    1280
#define CE      768
#define HEADS   8
#define HDIM    160
#define STXT    77
#define NIP     16
#define MROWS   (BATCH * SQ) // 28672

// ---------------------------------------------------------------------------
// Kernel 1: convert Wq / Wo (fp32 [K][N]) -> bf16 transposed [N][K]
// ---------------------------------------------------------------------------
__global__ __launch_bounds__(256) void wconv_kernel(
    const float* __restrict__ Wq, const float* __restrict__ Wo,
    unsigned short* __restrict__ WqT, unsigned short* __restrict__ WoT) {
    int i = blockIdx.x * 256 + threadIdx.x;
    const float* src = blockIdx.y ? Wo : Wq;
    unsigned short* dst = blockIdx.y ? WoT : WqT;
    int n = i / CDIM, k = i - n * CDIM;
    dst[(size_t)n * CDIM + k] = f2bf(src[(size_t)k * CDIM + n]);
}

// ---------------------------------------------------------------------------
// Kernel 1b: hidden_states fp32 -> bf16 (so the big GEMMs are pure-bf16 and
// can stream both tiles through the async LDS path)
// ---------------------------------------------------------------------------
__global__ __launch_bounds__(256) void h2bf_kernel(
    const float* __restrict__ x, unsigned short* __restrict__ y) {
    size_t i = ((size_t)blockIdx.x * 256 + threadIdx.x) * 4;
    float4 v = *(const float4*)&x[i];
    uint2 pk;
    pk.x = (unsigned)f2bf(v.x) | ((unsigned)f2bf(v.y) << 16);
    pk.y = (unsigned)f2bf(v.z) | ((unsigned)f2bf(v.w) << 16);
    *(uint2*)&y[i] = pk;
}

// ---------------------------------------------------------------------------
// Kernel 2: K/V projections for batches {0,6} (only ones the reference uses)
// ---------------------------------------------------------------------------
__global__ __launch_bounds__(256) void kvproj_kernel(
    const float* __restrict__ enc,
    const float* __restrict__ Wk, const float* __restrict__ Wv,
    const float* __restrict__ Wkip, const float* __restrict__ Wvip,
    unsigned short* __restrict__ Kenc, unsigned short* __restrict__ Vtenc,
    unsigned short* __restrict__ Kip,  unsigned short* __restrict__ Vtip) {
    const int row  = blockIdx.x;
    const int sel  = blockIdx.y;            // 0 -> batch 0, 1 -> batch 6
    const int kind = blockIdx.z;
    const int padded = (kind == 0) ? 80 : (kind == 1) ? 96 : 32;
    if (row >= padded) return;
    const bool isip  = (kind >= 2);
    const int  valid = isip ? NIP : STXT;
    const int  b     = sel ? 6 : 0;
    const int  tok   = isip ? (STXT + row) : row;
    const bool inr   = (row < valid);

    __shared__ float xs[CE];
    const int tid = threadIdx.x;
    if (inr)
        for (int i = tid; i < CE; i += 256)
            xs[i] = enc[((size_t)b * (STXT + NIP) + tok) * CE + i];
    __syncthreads();

    const float* W = (kind == 0) ? Wk : (kind == 1) ? Wv : (kind == 2) ? Wkip : Wvip;
    for (int c0 = 0; c0 < CDIM; c0 += 256) {
        int col = c0 + tid;
        float acc = 0.0f;
        if (inr)
            for (int k = 0; k < CE; ++k)
                acc = fmaf(xs[k], W[(size_t)k * CDIM + col], acc);
        unsigned short hv = f2bf(acc);
        if (kind == 0)      Kenc [(size_t)sel * 80  * CDIM + (size_t)row * CDIM + col] = hv;
        else if (kind == 1) Vtenc[(size_t)sel * CDIM * 96  + (size_t)col * 96  + row] = hv;
        else if (kind == 2) Kip  [(size_t)sel * 32  * CDIM + (size_t)row * CDIM + col] = hv;
        else                Vtip [(size_t)sel * CDIM * 32  + (size_t)col * 32  + row] = hv;
    }
}

// ---------------------------------------------------------------------------
// Kernel 3/5: double-buffered WMMA GEMM  C[M,N] = A[M,K] * Bt[N,K]^T
//   A, Bt bf16. Tiles stream global->LDS via async loads (ASYNCcnt) while the
//   previous tile's WMMAs run. Block 256 thr = 8 waves; tile 128x128;
//   wave tile 32x64; K-step 32.
// ---------------------------------------------------------------------------
#define LDS_STRIDE 40   // 32 k + 8 pad (80B rows, 16B aligned)

template <bool FINAL>
__global__ __launch_bounds__(256) void gemm_wmma(
    const unsigned short* __restrict__ A, const unsigned short* __restrict__ Bt,
    void* __restrict__ Cv, const float* __restrict__ bias,
    const float* __restrict__ resid, int N, int K) {
    __shared__ __align__(16) unsigned short As[2][128 * LDS_STRIDE];
    __shared__ __align__(16) unsigned short Bs[2][128 * LDS_STRIDE];

    const int tid = threadIdx.x;
    const int lane = tid & 31, wave = tid >> 5;
    const int wm = wave >> 1, wn = wave & 1;
    const int lr = lane & 15, khalf = lane >> 4;
    const int mBase = blockIdx.y * 128;
    const int nBase = blockIdx.x * 128;
    const int crow = tid >> 2, cg = (tid & 3) * 8;   // staging coords

    v8f acc[2][4];
#pragma unroll
    for (int f = 0; f < 2; ++f)
#pragma unroll
        for (int g = 0; g < 4; ++g) acc[f][g] = v8zero();

    auto stage = [&](int kb, int buf) {
#pragma unroll
        for (int p = 0; p < 2; ++p) {
            int row = p * 64 + crow;
            cp16(&As[buf][row * LDS_STRIDE + cg],
                 &A[(size_t)(mBase + row) * K + kb + cg]);
            cp16(&Bs[buf][row * LDS_STRIDE + cg],
                 &Bt[(size_t)(nBase + row) * K + kb + cg]);
        }
    };

    stage(0, 0);
    const int nk = K >> 5;
    for (int i = 0; i < nk; ++i) {
        lds_sync();                              // buf[i&1] ready for everyone
        if (i + 1 < nk) stage((i + 1) << 5, (i + 1) & 1);  // overlap next tile

        const unsigned short* Ab = As[i & 1];
        const unsigned short* Bb = Bs[i & 1];
        v16bf af[2], bg[4];
#pragma unroll
        for (int f = 0; f < 2; ++f)
            af[f] = ld_frag(&Ab[(wm * 32 + f * 16 + lr) * LDS_STRIDE], khalf);
#pragma unroll
        for (int g = 0; g < 4; ++g)
            bg[g] = ld_frag(&Bb[(wn * 64 + g * 16 + lr) * LDS_STRIDE], khalf);
#pragma unroll
        for (int f = 0; f < 2; ++f)
#pragma unroll
            for (int g = 0; g < 4; ++g)
                acc[f][g] = wmma_bf16(af[f], bg[g], acc[f][g]);
    }

#pragma unroll
    for (int f = 0; f < 2; ++f)
#pragma unroll
        for (int g = 0; g < 4; ++g) {
            int col = nBase + wn * 64 + g * 16 + lr;
            int rbase = mBase + wm * 32 + f * 16 + khalf * 8;
#pragma unroll
            for (int j = 0; j < 8; ++j) {
                float v = acc[f][g][j];
                size_t idx = (size_t)(rbase + j) * N + col;
                if (FINAL)
                    ((float*)Cv)[idx] = v + bias[col] + resid[idx];
                else
                    ((unsigned short*)Cv)[idx] = f2bf(v);
            }
        }
}

// ---------------------------------------------------------------------------
// Kernel 4: interpolated IP attention (async tile staging).
// ---------------------------------------------------------------------------
#define QS_STRIDE 168   // 160 d + 8 pad
#define VS_STRIDE 104   // 96 kv + 8 pad

__global__ __launch_bounds__(256) void attn_kernel(
    const unsigned short* __restrict__ Q,
    const unsigned short* __restrict__ Kenc, const unsigned short* __restrict__ Vtenc,
    const unsigned short* __restrict__ Kip,  const unsigned short* __restrict__ Vtip,
    unsigned short* __restrict__ O) {
    extern __shared__ __align__(16) unsigned short smem[];
    unsigned short* Qs  = smem;                       // [128][QS_STRIDE]
    unsigned short* Ks  = Qs + 128 * QS_STRIDE;       // [80][QS_STRIDE]
    unsigned short* Vts = Ks + 80 * QS_STRIDE;        // [160][VS_STRIDE]
    unsigned short* Ps  = Vts + 160 * VS_STRIDE;      // [8 waves][16][VS_STRIDE]

    const int tid = threadIdx.x;
    const int lane = tid & 31, wave = tid >> 5;
    const int lr = lane & 15, khalf = lane >> 4;
    const int qBase = blockIdx.x * 128;
    const int h = blockIdx.y;
    const int b = blockIdx.z;
    const float coef = (float)b * (1.0f / 6.0f);
    const float SCALE = 0.07905694150420949f;   // 1/sqrt(160)

    for (int c = tid; c < 128 * 20; c += 256) {
        int row = c / 20, g = c % 20;
        cp16(&Qs[row * QS_STRIDE + g * 8],
             &Q[((size_t)(b * SQ + qBase + row)) * CDIM + h * HDIM + g * 8]);
    }
    lds_sync();

    v16bf aQ[5];
    {
        const unsigned short* qrow = &Qs[(wave * 16 + lr) * QS_STRIDE];
#pragma unroll
        for (int kk = 0; kk < 5; ++kk) aQ[kk] = ld_frag(qrow + kk * 32, khalf);
    }

    v8f accO[10];
#pragma unroll
    for (int n = 0; n < 10; ++n) accO[n] = v8zero();

#pragma unroll 1
    for (int s = 0; s < 4; ++s) {
        const int  sel  = s >> 1;          // 0: batch0 k/v, 1: batch6 k/v
        const bool isip = (s & 1);
        const float wgt = sel ? coef : (1.0f - coef);
        const int kvlen = isip ? NIP : STXT;

        __syncthreads();   // everyone done reading previous Ks/Vts
        if (isip) {
            const unsigned short* Kg = Kip + (size_t)sel * 32 * CDIM;
            for (int c = tid; c < 32 * 20; c += 256) {
                int row = c / 20, g = c % 20;
                cp16(&Ks[row * QS_STRIDE + g * 8],
                     &Kg[(size_t)row * CDIM + h * HDIM + g * 8]);
            }
            const unsigned short* Vg = Vtip + (size_t)sel * CDIM * 32;
            for (int c = tid; c < 160 * 4; c += 256) {
                int row = c / 4, g = c % 4;
                cp16(&Vts[row * VS_STRIDE + g * 8],
                     &Vg[(size_t)(h * HDIM + row) * 32 + g * 8]);
            }
        } else {
            const unsigned short* Kg = Kenc + (size_t)sel * 80 * CDIM;
            for (int c = tid; c < 80 * 20; c += 256) {
                int row = c / 20, g = c % 20;
                cp16(&Ks[row * QS_STRIDE + g * 8],
                     &Kg[(size_t)row * CDIM + h * HDIM + g * 8]);
            }
            const unsigned short* Vg = Vtenc + (size_t)sel * CDIM * 96;
            for (int c = tid; c < 160 * 12; c += 256) {
                int row = c / 12, g = c % 12;
                cp16(&Vts[row * VS_STRIDE + g * 8],
                     &Vg[(size_t)(h * HDIM + row) * 96 + g * 8]);
            }
        }
        lds_sync();

        // logits = Q * K^T  (16 rows x 80 cols; stale cols masked below)
        v8f lg[5];
#pragma unroll
        for (int t = 0; t < 5; ++t) {
            lg[t] = v8zero();
            const unsigned short* krow = &Ks[(t * 16 + lr) * QS_STRIDE];
#pragma unroll
            for (int kk = 0; kk < 5; ++kk) {
                v16bf bk = ld_frag(krow + kk * 32, khalf);
                lg[t] = wmma_bf16(aQ[kk], bk, lg[t]);
            }
        }

        // masked softmax per row; blend weight folded into normalization
        unsigned short* prow0 = &Ps[(size_t)wave * 16 * VS_STRIDE];
#pragma unroll
        for (int j = 0; j < 8; ++j) {
            float vals[5];
            float mx = -3.0e38f;
#pragma unroll
            for (int t = 0; t < 5; ++t) {
                int col = t * 16 + lr;
                float v = (col < kvlen) ? lg[t][j] * SCALE : -3.0e38f;
                vals[t] = v;
                mx = fmaxf(mx, v);
            }
#pragma unroll
            for (int m = 1; m < 16; m <<= 1) mx = fmaxf(mx, __shfl_xor(mx, m, 32));
            float sum = 0.0f;
#pragma unroll
            for (int t = 0; t < 5; ++t) {
                float e = __expf(vals[t] - mx);
                vals[t] = e;
                sum += e;
            }
#pragma unroll
            for (int m = 1; m < 16; m <<= 1) sum += __shfl_xor(sum, m, 32);
            float inv = wgt / sum;
            unsigned short* pr = prow0 + (khalf * 8 + j) * VS_STRIDE;
#pragma unroll
            for (int t = 0; t < 5; ++t) pr[t * 16 + lr] = f2bf(vals[t] * inv);
            pr[80 + lr] = 0;   // zero kv pad 80..95
        }

        // PV accumulate: probs(16x96) x Vt -> acc(16x160)
#pragma unroll
        for (int ks = 0; ks < 3; ++ks) {
            v16bf aP = ld_frag(prow0 + lr * VS_STRIDE + ks * 32, khalf);
#pragma unroll
            for (int n = 0; n < 10; ++n) {
                v16bf bv = ld_frag(&Vts[(n * 16 + lr) * VS_STRIDE] + ks * 32, khalf);
                accO[n] = wmma_bf16(aP, bv, accO[n]);
            }
        }
    }

#pragma unroll
    for (int n = 0; n < 10; ++n) {
        int col = h * HDIM + n * 16 + lr;
        int rbase = qBase + wave * 16 + khalf * 8;
#pragma unroll
        for (int j = 0; j < 8; ++j)
            O[((size_t)(b * SQ + rbase + j)) * CDIM + col] = f2bf(accO[n][j]);
    }
}

// ---------------------------------------------------------------------------
// Launch
// ---------------------------------------------------------------------------
extern "C" void kernel_launch(void* const* d_in, const int* in_sizes, int n_in,
                              void* d_out, int out_size, void* d_ws, size_t ws_size,
                              hipStream_t stream) {
    (void)in_sizes; (void)n_in; (void)out_size; (void)ws_size;
    const float* hidden = (const float*)d_in[0];
    const float* enc    = (const float*)d_in[1];
    const float* Wq     = (const float*)d_in[2];
    const float* Wk     = (const float*)d_in[3];
    const float* Wv     = (const float*)d_in[4];
    const float* Wkip   = (const float*)d_in[5];
    const float* Wvip   = (const float*)d_in[6];
    const float* Wo     = (const float*)d_in[7];
    const float* bo     = (const float*)d_in[8];

    char* ws = (char*)d_ws;
    unsigned short* WqT    = (unsigned short*)(ws);              // 3,276,800
    unsigned short* WoT    = (unsigned short*)(ws + 3276800);    // 3,276,800
    unsigned short* KencB  = (unsigned short*)(ws + 6553600);    //   409,600
    unsigned short* VtencB = (unsigned short*)(ws + 6963200);    //   491,520
    unsigned short* KipB   = (unsigned short*)(ws + 7454720);    //   163,840
    unsigned short* VtipB  = (unsigned short*)(ws + 7618560);    //   163,840
    unsigned short* Qbuf   = (unsigned short*)(ws + 7782400);    // 73,400,320
    unsigned short* Obuf   = (unsigned short*)(ws + 81182720);   // 73,400,320
    unsigned short* Hb     = (unsigned short*)(ws + 154583040);  // 73,400,320

    wconv_kernel<<<dim3(6400, 2), 256, 0, stream>>>(Wq, Wo, WqT, WoT);
    h2bf_kernel<<<dim3((MROWS * CDIM) / 1024), 256, 0, stream>>>(hidden, Hb);
    kvproj_kernel<<<dim3(96, 2, 4), 256, 0, stream>>>(enc, Wk, Wv, Wkip, Wvip,
                                                      KencB, VtencB, KipB, VtipB);
    gemm_wmma<false><<<dim3(10, 224), 256, 0, stream>>>(
        Hb, WqT, Qbuf, nullptr, nullptr, CDIM, CDIM);

    size_t smem = (size_t)(128 * QS_STRIDE + 80 * QS_STRIDE + 160 * VS_STRIDE +
                           8 * 16 * VS_STRIDE) * sizeof(unsigned short); // ~126.8 KB
    attn_kernel<<<dim3(32, 8, 7), 256, smem, stream>>>(Qbuf, KencB, VtencB, KipB,
                                                       VtipB, Obuf);

    gemm_wmma<true><<<dim3(10, 224), 256, 0, stream>>>(
        Obuf, WoT, d_out, bo, hidden, CDIM, CDIM);
}
